// Decoder_14396730376943
// MI455X (gfx1250) — matrix operand order repaired and checked by reading, not compiled, not run on hardware
//
#include <hip/hip_runtime.h>
#include <math.h>

// ---------------------------------------------------------------------------
// Problem constants (from reference)
// ---------------------------------------------------------------------------
static constexpr int kB   = 64;    // batch
static constexpr int kT   = 101;   // time steps
static constexpr int kCIN = 7;
static constexpr int kH   = 128;   // hidden
static constexpr int kW   = 256;   // MLP width
static constexpr int kWL  = 4;
static constexpr int kC   = kCIN + 1;            // 8 path channels
static constexpr int kL   = kC + kC*(kC-1)/2;    // 36 logsig channels
static constexpr int kKW  = (kT-1)/kWL;          // 25 windows
static constexpr int kNK  = kKW + 1;             // 26 knots
static constexpr int kN2  = kH * kL;             // 4608 (GEMM2 N)
static constexpr int kSTEPS = kT - 1;            // 100 RK4 steps
static constexpr int kBLK = 16;                  // cooperating workgroups
static constexpr int kNS  = kN2 / kBLK;          // 288 N-columns per block (= 8*36)
static constexpr int kHS  = kH / kBLK;           // 8 h-rows produced per block

typedef __attribute__((ext_vector_type(16))) _Float16 v16h;
typedef __attribute__((ext_vector_type(2)))  _Float16 v2h;
typedef __attribute__((ext_vector_type(8)))  float    v8f;

// ---------------------------------------------------------------------------
// Workspace layout (bytes)
// ---------------------------------------------------------------------------
static constexpr size_t kOffW1T = 0;                                  // 65536
static constexpr size_t kOffW2T = 65536;                              // 2359296
static constexpr size_t kOffDX  = kOffW2T + (size_t)kN2*kW*2;         // 2764800
static constexpr size_t kOffKG  = kOffDX + (size_t)kSTEPS*3*kB*kL*4;  // 2*32KB
static constexpr size_t kOffCNT = kOffKG + (size_t)2*kB*kH*4;         // 4 (+pad)

// ---------------------------------------------------------------------------
// Prep: transpose + f16-convert weights (inputs are row-major [K][N])
// ---------------------------------------------------------------------------
__global__ void convert_w1(const float* __restrict__ W1, _Float16* __restrict__ w1t) {
    int idx = blockIdx.x * 256 + threadIdx.x;           // 256*128
    if (idx >= kW * kH) return;
    int n = idx >> 7, k = idx & 127;
    w1t[idx] = (_Float16)W1[k * kW + n];
}

__global__ void convert_w2(const float* __restrict__ W2, _Float16* __restrict__ w2t) {
    int idx = blockIdx.x * 256 + threadIdx.x;           // 4608*256
    if (idx >= kN2 * kW) return;
    int n = idx >> 8, k = idx & 255;
    w2t[idx] = (_Float16)W2[(size_t)k * kN2 + n];
}

__global__ void init_cnt(unsigned* cnt) { if (threadIdx.x == 0) *cnt = 0u; }

// ---------------------------------------------------------------------------
// Prep: log-signature knots + Hermite dX(s) table, one thread per batch elem.
// dxt[((step*3 + u)*64 + b)*36 + l],  u: 0 -> s=t0, 1 -> t0+0.5, 2 -> t0+1
// ---------------------------------------------------------------------------
__global__ void prep_knots(const float* __restrict__ inputs, float* __restrict__ dxt) {
    int b = threadIdx.x;
    if (b >= kB) return;
    const float* inp = inputs + (size_t)b * kT * kCIN;

    float knots[kNK][kL];
    float mslope[kKW][kL];
    for (int l = 0; l < kL; ++l) knots[0][l] = 0.0f;

    for (int k = 0; k < kKW; ++k) {
        float P[kC], lvl1[kC], S2[kC][kC];
        for (int i = 0; i < kC; ++i) { P[i] = 0.f; lvl1[i] = 0.f; }
        for (int i = 0; i < kC; ++i) for (int j = 0; j < kC; ++j) S2[i][j] = 0.f;
        for (int w = 0; w < kWL; ++w) {
            int t0 = k * kWL + w;
            float dd[kC];
            dd[0] = 1.0f;
            for (int c = 0; c < kCIN; ++c)
                dd[c+1] = inp[(t0+1)*kCIN + c] - inp[t0*kCIN + c];
            for (int i = 0; i < kC; ++i)
                for (int j = 0; j < kC; ++j)
                    S2[i][j] += P[i] * dd[j];
            for (int i = 0; i < kC; ++i) { lvl1[i] += dd[i]; P[i] += dd[i]; }
        }
        float ls[kL];
        for (int i = 0; i < kC; ++i) ls[i] = lvl1[i];
        int c2 = kC;
        for (int i = 0; i < kC; ++i)
            for (int j = i+1; j < kC; ++j)
                ls[c2++] = 0.5f * (S2[i][j] - S2[j][i]);
        for (int l = 0; l < kL; ++l) knots[k+1][l] = knots[k][l] + ls[l];
    }
    for (int j = 0; j < kKW; ++j)
        for (int l = 0; l < kL; ++l)
            mslope[j][l] = (knots[j+1][l] - knots[j][l]) * 0.25f;

    for (int step = 0; step < kSTEPS; ++step) {
        for (int u = 0; u < 3; ++u) {
            float s = (float)step + 0.5f * (float)u;
            int idx = (int)floorf(s * 0.25f);
            if (idx > kKW - 1) idx = kKW - 1;
            if (idx < 0) idx = 0;
            float tau = (s - 4.0f*(float)idx) * 0.25f;
            float t2 = tau*tau;
            float dh00 = 6.f*t2 - 6.f*tau;
            float dh10 = 3.f*t2 - 4.f*tau + 1.f;
            float dh11 = 3.f*t2 - 2.f*tau;
            float* dst = dxt + (((size_t)step*3 + u)*kB + b) * kL;
            int jm0 = (idx > 0) ? idx - 1 : 0;
            int jm1 = idx;
            for (int l = 0; l < kL; ++l) {
                float x0 = knots[idx][l], x1 = knots[idx+1][l];
                dst[l] = dh00*(x0 - x1)*0.25f + dh10*mslope[jm0][l]
                                              + dh11*mslope[jm1][l];
            }
        }
    }
}

// ---------------------------------------------------------------------------
// WMMA fragment loaders (CDNA5 16-bit layouts, wave32)
// ---------------------------------------------------------------------------
__device__ inline v16h load_a_frag(const _Float16* base, int m0, int ldk, int kk, int lane) {
    const _Float16* p = base + (size_t)(m0 + (lane & 15)) * ldk + kk + 16 * (lane >> 4);
    v16h a;
#pragma unroll
    for (int v = 0; v < 8; ++v) {
        int kb = 2 * v;
        v2h t = *(const v2h*)(p + kb);
        a[2*v]   = t.x;
        a[2*v+1] = t.y;
    }
    return a;
}

__device__ inline v16h load_b_frag(const _Float16* base, int n0, int ldk, int kk, int lane) {
    // [n][ldk] layout, 16 contiguous K halfs -> 2x b128
    return *(const v16h*)(base + (size_t)(n0 + (lane & 15)) * ldk + kk + 16 * (lane >> 4));
}

// ---------------------------------------------------------------------------
// Cooperative RK4 scan: 16 workgroups (one per WGP), each owns a 288-column
// slice of GEMM2 (exactly 8 h-channels), W2-slice + W1 pinned in LDS,
// z / RK4-accumulator in registers, k exchanged via double-buffered global
// buffer with a device-scope atomic barrier each stage.
// ---------------------------------------------------------------------------
__global__ __launch_bounds__(1024, 1)
void ncde_multi(const float* __restrict__ init,
                const float* __restrict__ b1,
                const float* __restrict__ b2,
                const _Float16* __restrict__ w1t,
                const _Float16* __restrict__ w2t,
                const float* __restrict__ dxt,
                float* __restrict__ kG,
                unsigned* __restrict__ cnt,
                float* __restrict__ out) {
    __shared__ _Float16 w2s[kNS * kW];      // 147456 B  W2^T slice, resident
    __shared__ _Float16 w1s[kW * kH];       //  65536 B  W1^T, resident
    __shared__ _Float16 hidb[kB * kW];      //  32768 B  relu(mlp1) f16
    __shared__ _Float16 zsb[kB * kH];       //  16384 B  stage input f16
    __shared__ float    kSl[kB * kHS];      //   2048 B  this block's k slice
    __shared__ float    dXs[kB * kL];       //   9216 B
    __shared__ float    b1s[kW];            //   1024 B
    __shared__ float    b2sl[kNS];          //   1152 B
                                            // total ~269 KB of 320 KB

    const int tid  = threadIdx.x;
    const int lane = tid & 31;
    const int wave = tid >> 5;
    const int c    = blockIdx.x;

    // ---- state in registers: z and RK4 accumulator (thread owns i = tid + j*1024)
    float zr[8], ar[8];
#pragma unroll
    for (int j = 0; j < 8; ++j) {
        int i = tid + j * 1024;
        float v = init[i];
        zr[j] = v; ar[j] = 0.f;
        zsb[i] = (_Float16)v;
    }
    if (c == 0) {
#pragma unroll
        for (int j = 0; j < 8; ++j) {
            int i = tid + j * 1024;
            out[(size_t)(i >> 7) * (kT * kH) + (i & 127)] = tanhf(zr[j]);
        }
    }

    // ---- async bulk copy: this block's W2^T slice -> LDS (ASYNCcnt path)
    {
        const _Float16* src = w2t + (size_t)c * kNS * kW;
        for (int e = tid * 16; e < kNS * kW; e += 1024 * 16) {
            unsigned ldsa = (unsigned)(uintptr_t)&w2s[e];
            unsigned goff = (unsigned)(e * 2);
            asm volatile("global_load_async_to_lds_b128 %0, %1, %2 offset:0"
                         :: "v"(ldsa), "v"(goff), "s"(src) : "memory");
        }
    }
    for (int i = tid; i < kW * kH; i += 1024) w1s[i] = w1t[i];
    for (int i = tid; i < kW;  i += 1024) b1s[i] = b1[i];
    for (int i = tid; i < kNS; i += 1024) b2sl[i] = b2[c * kNS + i];
    for (int i = tid; i < kB * kL; i += 1024) dXs[i] = dxt[i];   // step0/stage0
    for (int i = tid; i < kB * kHS; i += 1024) kSl[i] = 0.f;
    asm volatile("s_wait_asynccnt 0" ::: "memory");
    __syncthreads();

    const float coef[4] = {0.f, 0.5f, 0.5f, 1.f};
    const float wgt[4]  = {1.f, 2.f, 2.f, 1.f};
    const int   uidx[4] = {0, 1, 1, 2};

    for (int sidx = 0; sidx < 4 * kSTEPS; ++sidx) {
        const int st = sidx & 3;

        // P1: GEMM1 hid = relu(y @ W1 + b1), all operands in LDS (redundant/block)
        for (int t = wave; t < (kB/16) * (kW/16); t += 32) {
            int m0 = (t >> 4) * 16, n0 = (t & 15) * 16;
            v8f cc = {};
#pragma unroll
            for (int kk = 0; kk < kH; kk += 32) {
                v16h a  = load_a_frag(zsb, m0, kH, kk, lane);
                v16h bf = load_b_frag(w1s, n0, kH, kk, lane);
                cc = __builtin_amdgcn_wmma_f32_16x16x32_f16(
                        false, a, false, bf, (short)0, cc, false, false);
            }
            int n  = n0 + (lane & 15);
            int mh = (lane >> 4) * 8;
#pragma unroll
            for (int r = 0; r < 8; ++r) {
                float hv = cc[r] + b1s[n];
                hidb[(size_t)(m0 + mh + r) * kW + n] = (_Float16)(hv > 0.f ? hv : 0.f);
            }
        }
        __syncthreads();

        // P2: GEMM2 slice + tanh + dX contraction into kSl (all LDS)
        for (int t = wave; t < (kB/16) * (kNS/16); t += 32) {
            int mt = t / (kNS/16), nt = t % (kNS/16);
            int m0 = mt * 16;
            const _Float16* bb = w2s + (size_t)(nt*16 + (lane & 15)) * kW + 16 * (lane >> 4);
            v8f cc = {};
#pragma unroll
            for (int kk = 0; kk < kW; kk += 32) {
                v16h a  = load_a_frag(hidb, m0, kW, kk, lane);
                v16h bf = *(const v16h*)(bb + kk);
                cc = __builtin_amdgcn_wmma_f32_16x16x32_f16(
                        false, a, false, bf, (short)0, cc, false, false);
            }
            int nl = nt*16 + (lane & 15);              // local column in slice
            int hl = nl / kL, l = nl % kL;             // slice covers h = c*8+hl
            int mh = (lane >> 4) * 8;
#pragma unroll
            for (int r = 0; r < 8; ++r) {
                int m = m0 + mh + r;                   // batch row
                float val = tanhf(cc[r] + b2sl[nl]);
                atomicAdd(&kSl[m * kHS + hl], val * dXs[m * kL + l]);
            }
        }
        __syncthreads();

        // P3: publish k slice -> kG[parity], then device-scope barrier
        float* kbuf = kG + (size_t)(sidx & 1) * kB * kH;
        if (tid < kB * kHS) {
            int b = tid >> 3, hl = tid & 7;
            kbuf[b * kH + c * kHS + hl] = kSl[tid];
        }
        __threadfence();                     // make slice visible (release)
        __syncthreads();
        if (tid == 0) {
            __hip_atomic_fetch_add(cnt, 1u, __ATOMIC_RELEASE, __HIP_MEMORY_SCOPE_AGENT);
            unsigned expected = (unsigned)(kBLK * (sidx + 1));
            while (__hip_atomic_load(cnt, __ATOMIC_ACQUIRE, __HIP_MEMORY_SCOPE_AGENT)
                   < expected)
                __builtin_amdgcn_s_sleep(1);
        }
        __syncthreads();
        __threadfence();                     // acquire: invalidate stale kG lines

        // P4: consume full k, fold RK4, build next stage input; prep next dX
        const float wg = wgt[st];
#pragma unroll
        for (int j = 0; j < 8; ++j) {
            int i = tid + j * 1024;
            float kv = kbuf[i];
            ar[j] += wg * kv;
            if (st < 3) {
                zsb[i] = (_Float16)(zr[j] + coef[st + 1] * kv);
            } else {
                float zn = zr[j] + ar[j] * (1.0f / 6.0f);
                zr[j] = zn; ar[j] = 0.f;
                zsb[i] = (_Float16)zn;
                if (c == 0)
                    out[(size_t)(i >> 7) * (kT * kH)
                        + (size_t)((sidx >> 2) + 1) * kH + (i & 127)] = tanhf(zn);
            }
        }
        for (int i = tid; i < kB * kHS; i += 1024) kSl[i] = 0.f;
        if (sidx + 1 < 4 * kSTEPS) {
            int sn = sidx + 1;
            const float* dsrc = dxt + (((size_t)(sn >> 2) * 3 + uidx[sn & 3]) * kB) * kL;
            for (int i = tid; i < kB * kL; i += 1024) dXs[i] = dsrc[i];
        }
        __syncthreads();
    }
}

// ---------------------------------------------------------------------------
// Launch
// ---------------------------------------------------------------------------
extern "C" void kernel_launch(void* const* d_in, const int* in_sizes, int n_in,
                              void* d_out, int out_size, void* d_ws, size_t ws_size,
                              hipStream_t stream) {
    const float* init   = (const float*)d_in[0];
    const float* inputs = (const float*)d_in[1];
    const float* W1     = (const float*)d_in[2];
    const float* b1     = (const float*)d_in[3];
    const float* W2     = (const float*)d_in[4];
    const float* b2     = (const float*)d_in[5];
    float* out = (float*)d_out;

    char* ws = (char*)d_ws;
    _Float16* w1t = (_Float16*)(ws + kOffW1T);
    _Float16* w2t = (_Float16*)(ws + kOffW2T);
    float*    dxt = (float*)   (ws + kOffDX);
    float*    kG  = (float*)   (ws + kOffKG);
    unsigned* cnt = (unsigned*)(ws + kOffCNT);

    convert_w1<<<(kW * kH + 255) / 256, 256, 0, stream>>>(W1, w1t);
    convert_w2<<<(kN2 * kW + 255) / 256, 256, 0, stream>>>(W2, w2t);
    prep_knots<<<1, 64, 0, stream>>>(inputs, dxt);
    init_cnt<<<1, 1, 0, stream>>>(cnt);
    ncde_multi<<<kBLK, 1024, 0, stream>>>(init, b1, b2, w1t, w2t, dxt, kG, cnt, out);
}
